// EGNNDynamicsTransferableMD_87892210746068
// MI455X (gfx1250) — matrix operand order repaired and checked by reading, not compiled, not run on hardware
//
#include <hip/hip_runtime.h>
#include <hip/hip_bf16.h>

typedef __attribute__((ext_vector_type(16))) _Float16 v16h;
typedef __attribute__((ext_vector_type(8)))  float    v8f;

#define NB   16
#define NN   128
#define CC   128
#define NNODE 2048   // NB*NN

__device__ __forceinline__ float sigm_f(float x) { return 1.0f / (1.0f + __expf(-x)); }
__device__ __forceinline__ float silu_f(float x) { return x * sigm_f(x); }

// ---- WMMA wrapper (CDNA5: D = A(16x32 f16) * B(32x16 f16) + C(16x16 f32)) ----
__device__ __forceinline__ v8f wmma32(v16h a, v16h b, v8f c) {
  return __builtin_amdgcn_wmma_f32_16x16x32_f16(
      /*neg_a=*/false, a, /*neg_b=*/false, b,
      /*c_mod=*/(short)0, c, /*reuse_a=*/false, /*reuse_b=*/false);
}

// A operand loader: tile rows are M (16), cols K (32), row-major with stride
// `stride` halves. Lane layout: contiguous k = 16*half + e (matches B fragments):
// one contiguous 32-byte run per lane -> compiler emits 2x ds_load_b128.
__device__ __forceinline__ v16h load_A_lds(const _Float16* base, int stride,
                                           int kBase, int lane) {
  int m = lane & 15, half = lane >> 4;
  const _Float16* row = base + m * stride + kBase;
  v16h a;
#pragma unroll
  for (int i = 0; i < 8; ++i) {
    int k0 = 16 * half + 2 * i;
    a[2 * i]     = row[k0];
    a[2 * i + 1] = row[k0 + 1];
  }
  return a;
}

// B operand loader from a pre-swizzled buffer: each lane's fragment for
// (kb, ct) is 16 contiguous halves (32 bytes) -> wide b128 loads.
// Fragment order: ((kb*8 + ct)*32 + lane)*16. N is fixed at 128 (8 col tiles).
__device__ __forceinline__ v16h load_frag(const _Float16* base, int kb, int ct,
                                          int lane) {
  return *(const v16h*)(base + (((size_t)(kb * 8 + ct) * 32 + lane) << 4));
}

// ---------------------------------------------------------------------------
// Prep: one-hot embedding GEMM collapses to a gather-sum of emb_W rows.
// ---------------------------------------------------------------------------
__global__ void prep_kernel(const float* __restrict__ t, const float* __restrict__ x,
                            const int* __restrict__ atomt, const int* __restrict__ aapos,
                            const int* __restrict__ aatyp, const int* __restrict__ seql,
                            const float* __restrict__ nmask,
                            const float* __restrict__ embW, const float* __restrict__ embb,
                            float* __restrict__ hF32, _Float16* __restrict__ hF16,
                            float* __restrict__ x0, float* __restrict__ xcur) {
  int node = blockIdx.x;
  int b = node >> 7, nl = node & 127;
  int c = threadIdx.x;
  float mask = nmask[node];
  int a = atomt[node] - 1; if (a < 0) a = 0;
  int p = aapos[node] - 1; if (p < 0) p = 0;
  int q = aatyp[node] - 1; if (q < 0) q = 0;
  int se = seql[b] - 5;    if (se < 0) se = 0;
  float v = embW[(size_t)a * CC + c] + embW[(size_t)(54 + p) * CC + c] +
            embW[(size_t)(84 + q) * CC + c];
  if (se < 26) v += embW[(size_t)(104 + se) * CC + c];
  v += t[b] * embW[(size_t)130 * CC + c];
  v = mask * v + embb[c];
  hF32[(size_t)node * CC + c] = v;
  hF16[(size_t)node * CC + c] = (_Float16)v;
  if (c < 3) {
    float xv = x[(size_t)b * (NN * 3) + nl * 3 + c] * mask;
    x0[node * 3 + c] = xv;
    xcur[node * 3 + c] = xv;
  }
}

// ---------------------------------------------------------------------------
// Convert f32 weight (K x 128 row-major) to f16 in WMMA B-fragment order.
// dst[((kb*8+ct)*32+lane)*16 + e] = src[(kb*32 + 16*(lane>>4) + e)*128 + ct*16 + (lane&15)]
// ---------------------------------------------------------------------------
__global__ void cvt_swz(const float* __restrict__ src, _Float16* __restrict__ dst,
                        int K) {
  int o = blockIdx.x * 256 + threadIdx.x;
  if (o >= K * 128) return;
  int f = o >> 9;            // kb*8 + ct
  int r = o & 511;
  int lane = r >> 4, e = r & 15;
  int kb = f >> 3, ct = f & 7;
  int k = kb * 32 + ((lane >> 4) << 4) + e;
  int n = (ct << 4) + (lane & 15);
  dst[o] = (_Float16)src[(size_t)k * 128 + n];
}

// ---------------------------------------------------------------------------
// P/Q kernel: P = H @ We1[:,0:128], Q = H @ We1[:,128:256]   (per layer)
// grid 128 blocks (16 nodes each) x 256 threads (8 waves).
// ---------------------------------------------------------------------------
__global__ __launch_bounds__(256) void pq_kernel(const _Float16* __restrict__ hF16,
                                                 const _Float16* __restrict__ W1aS,
                                                 const _Float16* __restrict__ W1bS,
                                                 float* __restrict__ P,
                                                 float* __restrict__ Q) {
  __shared__ alignas(32) _Float16 shA[16 * CC];
  int tid = threadIdx.x;
  int nb = blockIdx.x;
  {
    const v16h* s = (const v16h*)(hF16 + (size_t)nb * 16 * CC);
    v16h* d = (v16h*)shA;
    for (int idx = tid; idx < (16 * CC) / 16; idx += 256) d[idx] = s[idx];
  }
  __syncthreads();

  int wv = tid >> 5, lane = tid & 31, half = lane >> 4, nIdx = lane & 15;
  v8f accP, accQ;
#pragma unroll
  for (int r = 0; r < 8; ++r) { accP[r] = 0.f; accQ[r] = 0.f; }
#pragma unroll
  for (int kb = 0; kb < 4; ++kb) {
    v16h a  = load_A_lds(shA, CC, kb * 32, lane);
    v16h bP = load_frag(W1aS, kb, wv, lane);
    v16h bQ = load_frag(W1bS, kb, wv, lane);
    accP = wmma32(a, bP, accP);
    accQ = wmma32(a, bQ, accQ);
  }
#pragma unroll
  for (int r = 0; r < 8; ++r) {
    int row = r + 8 * half;
    size_t g = (size_t)(nb * 16 + row) * CC + wv * 16 + nIdx;
    P[g] = accP[r];
    Q[g] = accQ[r];
  }
}

// ---------------------------------------------------------------------------
// Edge kernel: one workgroup per destination node (b,i); its 128 edges (all j).
// Weights staged once per WG into LDS (swizzled fragments -> ds_load_b128).
// grid 2048 x 256 threads (8 waves).
// ---------------------------------------------------------------------------
__global__ __launch_bounds__(256) void edge_kernel(
    const float* __restrict__ xcur, const float* __restrict__ x0,
    const float* __restrict__ nmask,
    const float* __restrict__ P, const float* __restrict__ Q,
    const float* __restrict__ be1, const float* __restrict__ w_r,
    const float* __restrict__ w_e,
    const _Float16* __restrict__ We2s, const float* __restrict__ be2,
    const float* __restrict__ Wa, const float* __restrict__ ba,
    const _Float16* __restrict__ Wc1s, const float* __restrict__ bc1,
    const float* __restrict__ Wc2,
    float* __restrict__ agg, float* __restrict__ dx) {
  __shared__ alignas(32) _Float16 shm[NN * CC];   // m1, then m (in-place per wave)
  __shared__ alignas(32) _Float16 shW[CC * CC];   // staged weights (We2 then Wc1)
  __shared__ float sP[CC], sWr[CC], sWe[CC], sBe2[CC], sBc1[CC], sWa[CC], sWc2[CC];
  __shared__ float sdiff[3 * NN], srad[NN], sea[NN], smask[NN];
  __shared__ float gacc[NN], tacc[NN];
  __shared__ float redv[3 * NN];
  __shared__ float aggp[256];

  int tid = threadIdx.x;
  int node = blockIdx.x;
  int b = node >> 7, i = node & 127;
  const float* Qb = Q + (size_t)b * NN * CC;

  if (tid < NN) {
    int j = tid;
    int nj = b * NN + j;
    float xj0 = xcur[nj * 3 + 0], xj1 = xcur[nj * 3 + 1], xj2 = xcur[nj * 3 + 2];
    float xi0 = xcur[node * 3 + 0], xi1 = xcur[node * 3 + 1], xi2 = xcur[node * 3 + 2];
    float d0 = xi0 - xj0, d1 = xi1 - xj1, d2 = xi2 - xj2;
    sdiff[j] = d0; sdiff[NN + j] = d1; sdiff[2 * NN + j] = d2;
    srad[j] = d0 * d0 + d1 * d1 + d2 * d2;
    float e0 = x0[node * 3 + 0] - x0[nj * 3 + 0];
    float e1 = x0[node * 3 + 1] - x0[nj * 3 + 1];
    float e2 = x0[node * 3 + 2] - x0[nj * 3 + 2];
    sea[j] = e0 * e0 + e1 * e1 + e2 * e2;
    smask[j] = nmask[nj];
    gacc[j] = 0.f; tacc[j] = 0.f;
    sP[j]  = P[(size_t)node * CC + j] + be1[j];
    sWr[j] = w_r[j]; sWe[j] = w_e[j];
    sBe2[j] = be2[j]; sBc1[j] = bc1[j];
    sWa[j] = Wa[j];  sWc2[j] = Wc2[j];
  }
  // stage We2 fragments into LDS (wide 32B copies)
  {
    const v16h* s = (const v16h*)We2s;
    v16h* d = (v16h*)shW;
    for (int idx = tid; idx < (CC * CC) / 16; idx += 256) d[idx] = s[idx];
  }
  __syncthreads();

  // m1 = silu(P[i] + Q[j] + radial*w_r + ea*w_e + be1), float4-vectorized
  {
    const float4* Q4 = (const float4*)Qb;
    for (int idx = tid; idx < (NN * CC) / 4; idx += 256) {
      int j = idx >> 5;
      int c = (idx & 31) << 2;
      float4 qv = Q4[idx];
      float rj = srad[j], ej = sea[j];
      shm[(j << 7) + c + 0] = (_Float16)silu_f(sP[c + 0] + qv.x + rj * sWr[c + 0] + ej * sWe[c + 0]);
      shm[(j << 7) + c + 1] = (_Float16)silu_f(sP[c + 1] + qv.y + rj * sWr[c + 1] + ej * sWe[c + 1]);
      shm[(j << 7) + c + 2] = (_Float16)silu_f(sP[c + 2] + qv.z + rj * sWr[c + 2] + ej * sWe[c + 2]);
      shm[(j << 7) + c + 3] = (_Float16)silu_f(sP[c + 3] + qv.w + rj * sWr[c + 3] + ej * sWe[c + 3]);
    }
  }
  __syncthreads();

  int wv = tid >> 5, lane = tid & 31, half = lane >> 4, nIdx = lane & 15;
  int mBase = wv * 16;

  // GEMM2: m2 = silu(m1 @ We2 + be2); all 8 col tiles kept in registers so m1
  // rows (owned exclusively by this wave) can be overwritten in place.
  {
    v8f acc[8];
#pragma unroll
    for (int ct = 0; ct < 8; ++ct)
#pragma unroll
      for (int r = 0; r < 8; ++r) acc[ct][r] = sBe2[ct * 16 + nIdx];
#pragma unroll
    for (int kb = 0; kb < 4; ++kb) {
      v16h a = load_A_lds(shm + mBase * CC, CC, kb * 32, lane);
#pragma unroll
      for (int ct = 0; ct < 8; ++ct) {
        v16h bb = load_frag(shW, kb, ct, lane);
        acc[ct] = wmma32(a, bb, acc[ct]);
      }
    }
    float gpart[8];
#pragma unroll
    for (int r = 0; r < 8; ++r) gpart[r] = 0.f;
#pragma unroll
    for (int ct = 0; ct < 8; ++ct) {
#pragma unroll
      for (int r = 0; r < 8; ++r) {
        float s = silu_f(acc[ct][r]);
        int row = mBase + half * 8 + r;
        int col = ct * 16 + nIdx;
        shm[row * CC + col] = (_Float16)s;
        gpart[r] += s * sWa[col];
      }
    }
#pragma unroll
    for (int r = 0; r < 8; ++r)
      atomicAdd(&gacc[mBase + half * 8 + r], gpart[r]);
  }
  __syncthreads();

  // gate + edge mask; concurrently stage Wc1 fragments (shW free after barrier)
  {
    const v16h* s = (const v16h*)Wc1s;
    v16h* d = (v16h*)shW;
    for (int idx = tid; idx < (CC * CC) / 16; idx += 256) d[idx] = s[idx];
  }
  {
    float mi = smask[i];
    float ba0 = ba[0];
    for (int idx = tid; idx < NN * CC; idx += 256) {
      int j = idx >> 7;
      float g = sigm_f(gacc[j] + ba0);
      float em = mi * smask[j] * ((j == i) ? 0.f : 1.f);
      shm[idx] = (_Float16)((float)shm[idx] * g * em);
    }
  }
  __syncthreads();

  // GEMM3: c = silu(m @ Wc1 + bc1); fuse t = c . Wc2 (c never stored).
  {
    v8f acc[8];
#pragma unroll
    for (int ct = 0; ct < 8; ++ct)
#pragma unroll
      for (int r = 0; r < 8; ++r) acc[ct][r] = sBc1[ct * 16 + nIdx];
#pragma unroll
    for (int kb = 0; kb < 4; ++kb) {
      v16h a = load_A_lds(shm + mBase * CC, CC, kb * 32, lane);
#pragma unroll
      for (int ct = 0; ct < 8; ++ct) {
        v16h bb = load_frag(shW, kb, ct, lane);
        acc[ct] = wmma32(a, bb, acc[ct]);
      }
    }
    float tpart[8];
#pragma unroll
    for (int r = 0; r < 8; ++r) tpart[r] = 0.f;
#pragma unroll
    for (int ct = 0; ct < 8; ++ct)
#pragma unroll
      for (int r = 0; r < 8; ++r)
        tpart[r] += silu_f(acc[ct][r]) * sWc2[ct * 16 + nIdx];
#pragma unroll
    for (int r = 0; r < 8; ++r)
      atomicAdd(&tacc[mBase + half * 8 + r], tpart[r]);
  }
  __syncthreads();

  // trans_j = diff_j * tanh(t_j); row-local segment sums (dx, agg)
  if (tid < NN) {
    int j = tid;
    float th = tanhf(tacc[j]);
    redv[j] = sdiff[j] * th;
    redv[NN + j] = sdiff[NN + j] * th;
    redv[2 * NN + j] = sdiff[2 * NN + j] * th;
  }
  {
    int c = tid & 127, hh = tid >> 7;
    float s = 0.f;
    for (int j = hh * 64; j < hh * 64 + 64; ++j)
      s += (float)shm[j * CC + c];
    aggp[tid] = s;
  }
  __syncthreads();
  if (tid < NN)
    agg[(size_t)node * CC + tid] = aggp[tid] + aggp[128 + tid];
  for (int st = 64; st >= 1; st >>= 1) {
    if (tid < st) {
      redv[tid] += redv[tid + st];
      redv[NN + tid] += redv[NN + tid + st];
      redv[2 * NN + tid] += redv[2 * NN + tid + st];
    }
    __syncthreads();
  }
  if (tid < 3) dx[node * 3 + tid] = redv[tid * NN];
}

// ---------------------------------------------------------------------------
// Node kernel: x update + hn = silu([h,agg] @ Wn1 + bn1) @ Wn2 + bn2; h+=hn.
// grid 128 blocks (16 nodes) x 256 threads (8 waves, col tile = wave).
// ---------------------------------------------------------------------------
__global__ __launch_bounds__(256) void node_kernel(
    float* __restrict__ xcur, const float* __restrict__ dx,
    const float* __restrict__ nmask,
    float* __restrict__ hF32, _Float16* __restrict__ hF16,
    const float* __restrict__ agg,
    const _Float16* __restrict__ Wn1s, const float* __restrict__ bn1,
    const _Float16* __restrict__ Wn2s, const float* __restrict__ bn2) {
  __shared__ alignas(32) _Float16 shA[16 * 256];
  __shared__ alignas(32) _Float16 shA2[16 * CC];
  int tid = threadIdx.x;
  int nb = blockIdx.x;

  if (tid < 48) {
    int node = nb * 16 + tid / 3, d = tid % 3;
    float m = nmask[node];
    xcur[node * 3 + d] = (xcur[node * 3 + d] + dx[node * 3 + d]) * m;
  }
  for (int idx = tid; idx < 16 * 256; idx += 256) {
    int row = idx >> 8, c = idx & 255;
    int node = nb * 16 + row;
    float v = (c < CC) ? hF32[(size_t)node * CC + c]
                       : agg[(size_t)node * CC + (c - CC)];
    shA[idx] = (_Float16)v;
  }
  __syncthreads();

  int wv = tid >> 5, lane = tid & 31, half = lane >> 4, nIdx = lane & 15;
  {
    v8f acc;
#pragma unroll
    for (int r = 0; r < 8; ++r) acc[r] = bn1[wv * 16 + nIdx];
#pragma unroll
    for (int kb = 0; kb < 8; ++kb) {
      v16h a = load_A_lds(shA, 256, kb * 32, lane);
      v16h bb = load_frag(Wn1s, kb, wv, lane);
      acc = wmma32(a, bb, acc);
    }
#pragma unroll
    for (int r = 0; r < 8; ++r) {
      int row = r + 8 * half;
      shA2[row * CC + wv * 16 + nIdx] = (_Float16)silu_f(acc[r]);
    }
  }
  __syncthreads();
  {
    v8f acc;
#pragma unroll
    for (int r = 0; r < 8; ++r) acc[r] = bn2[wv * 16 + nIdx];
#pragma unroll
    for (int kb = 0; kb < 4; ++kb) {
      v16h a = load_A_lds(shA2, CC, kb * 32, lane);
      v16h bb = load_frag(Wn2s, kb, wv, lane);
      acc = wmma32(a, bb, acc);
    }
#pragma unroll
    for (int r = 0; r < 8; ++r) {
      int row = r + 8 * half;
      int node = nb * 16 + row;
      int col = wv * 16 + nIdx;
      float m = nmask[node];
      float hv = (hF32[(size_t)node * CC + col] + acc[r]) * m;
      hF32[(size_t)node * CC + col] = hv;
      hF16[(size_t)node * CC + col] = (_Float16)hv;
    }
  }
}

// ---------------------------------------------------------------------------
// Final: vel = (xcur - x0)*mask, subtract per-batch masked mean.
// ---------------------------------------------------------------------------
__global__ void final_kernel(const float* __restrict__ xcur, const float* __restrict__ x0,
                             const float* __restrict__ nmask, float* __restrict__ out) {
  __shared__ float s0[NN], s1[NN], s2[NN], sc[NN];
  int b = blockIdx.x, n = threadIdx.x;
  int node = b * NN + n;
  float m = nmask[node];
  float v0 = (xcur[node * 3 + 0] - x0[node * 3 + 0]) * m;
  float v1 = (xcur[node * 3 + 1] - x0[node * 3 + 1]) * m;
  float v2 = (xcur[node * 3 + 2] - x0[node * 3 + 2]) * m;
  s0[n] = v0; s1[n] = v1; s2[n] = v2; sc[n] = m;
  __syncthreads();
  for (int st = 64; st >= 1; st >>= 1) {
    if (n < st) { s0[n] += s0[n + st]; s1[n] += s1[n + st]; s2[n] += s2[n + st]; sc[n] += sc[n + st]; }
    __syncthreads();
  }
  float inv = 1.0f / sc[0];
  out[(size_t)b * (NN * 3) + n * 3 + 0] = v0 - s0[0] * inv * m;
  out[(size_t)b * (NN * 3) + n * 3 + 1] = v1 - s1[0] * inv * m;
  out[(size_t)b * (NN * 3) + n * 3 + 2] = v2 - s2[0] * inv * m;
}

extern "C" void kernel_launch(void* const* d_in, const int* in_sizes, int n_in,
                              void* d_out, int out_size, void* d_ws, size_t ws_size,
                              hipStream_t stream) {
  (void)in_sizes; (void)n_in; (void)out_size; (void)ws_size;
  const float* t     = (const float*)d_in[0];
  const float* x     = (const float*)d_in[1];
  const int*   atomt = (const int*)d_in[2];
  const int*   aapos = (const int*)d_in[3];
  const int*   aatyp = (const int*)d_in[4];
  const int*   seql  = (const int*)d_in[5];
  const float* nmask = (const float*)d_in[6];
  const float* embW  = (const float*)d_in[7];
  const float* embb  = (const float*)d_in[8];
  const float* We1   = (const float*)d_in[9];
  const float* be1   = (const float*)d_in[10];
  const float* We2   = (const float*)d_in[11];
  const float* be2   = (const float*)d_in[12];
  const float* Wa    = (const float*)d_in[13];
  const float* ba    = (const float*)d_in[14];
  const float* Wn1   = (const float*)d_in[15];
  const float* bn1   = (const float*)d_in[16];
  const float* Wn2   = (const float*)d_in[17];
  const float* bn2   = (const float*)d_in[18];
  const float* Wc1   = (const float*)d_in[19];
  const float* bc1   = (const float*)d_in[20];
  const float* Wc2   = (const float*)d_in[21];

  char* ws = (char*)d_ws;
  size_t off = 0;
  auto take = [&](size_t bytes) -> char* {
    off = (off + 255) & ~(size_t)255;
    char* p = ws + off;
    off += bytes;
    return p;
  };
  float*    hF32  = (float*)take((size_t)NNODE * CC * 4);
  _Float16* hF16  = (_Float16*)take((size_t)NNODE * CC * 2);
  float*    Pbuf  = (float*)take((size_t)NNODE * CC * 4);
  float*    Qbuf  = (float*)take((size_t)NNODE * CC * 4);
  float*    aggB  = (float*)take((size_t)NNODE * CC * 4);
  float*    x0    = (float*)take((size_t)NNODE * 3 * 4);
  float*    xcur  = (float*)take((size_t)NNODE * 3 * 4);
  float*    dxB   = (float*)take((size_t)NNODE * 3 * 4);
  _Float16* We1aS = (_Float16*)take((size_t)3 * CC * CC * 2);   // rows 0..127
  _Float16* We1bS = (_Float16*)take((size_t)3 * CC * CC * 2);   // rows 128..255
  _Float16* We2S  = (_Float16*)take((size_t)3 * CC * CC * 2);
  _Float16* Wc1S  = (_Float16*)take((size_t)3 * CC * CC * 2);
  _Float16* Wn1S  = (_Float16*)take((size_t)3 * 256 * CC * 2);
  _Float16* Wn2S  = (_Float16*)take((size_t)3 * CC * CC * 2);

  prep_kernel<<<NNODE, 128, 0, stream>>>(t, x, atomt, aapos, aatyp, seql, nmask,
                                         embW, embb, hF32, hF16, x0, xcur);
  // Swizzle weights into WMMA B-fragment order (per layer slices).
  for (int l = 0; l < 3; ++l) {
    const int nk1 = CC * 128, gk1 = (nk1 + 255) / 256;
    cvt_swz<<<gk1, 256, 0, stream>>>(We1 + (size_t)l * 258 * CC,
                                     We1aS + (size_t)l * CC * CC, 128);
    cvt_swz<<<gk1, 256, 0, stream>>>(We1 + (size_t)l * 258 * CC + 128 * CC,
                                     We1bS + (size_t)l * CC * CC, 128);
    cvt_swz<<<gk1, 256, 0, stream>>>(We2 + (size_t)l * CC * CC,
                                     We2S + (size_t)l * CC * CC, 128);
    cvt_swz<<<gk1, 256, 0, stream>>>(Wc1 + (size_t)l * CC * CC,
                                     Wc1S + (size_t)l * CC * CC, 128);
    const int nk2 = 256 * 128, gk2 = (nk2 + 255) / 256;
    cvt_swz<<<gk2, 256, 0, stream>>>(Wn1 + (size_t)l * 256 * CC,
                                     Wn1S + (size_t)l * 256 * CC, 256);
    cvt_swz<<<gk1, 256, 0, stream>>>(Wn2 + (size_t)l * CC * CC,
                                     Wn2S + (size_t)l * CC * CC, 128);
  }

  for (int l = 0; l < 3; ++l) {
    pq_kernel<<<128, 256, 0, stream>>>(hF16, We1aS + (size_t)l * CC * CC,
                                       We1bS + (size_t)l * CC * CC, Pbuf, Qbuf);
    edge_kernel<<<NNODE, 256, 0, stream>>>(
        xcur, x0, nmask, Pbuf, Qbuf,
        be1 + l * CC,
        We1 + (size_t)l * 258 * CC + 256 * CC,   // radial weight row
        We1 + (size_t)l * 258 * CC + 257 * CC,   // edge_attr weight row
        We2S + (size_t)l * CC * CC, be2 + l * CC,
        Wa + l * CC, ba + l,
        Wc1S + (size_t)l * CC * CC, bc1 + l * CC, Wc2 + l * CC,
        aggB, dxB);
    node_kernel<<<128, 256, 0, stream>>>(xcur, dxB, nmask, hF32, hF16, aggB,
                                         Wn1S + (size_t)l * 256 * CC, bn1 + l * CC,
                                         Wn2S + (size_t)l * CC * CC, bn2 + l * CC);
  }
  final_kernel<<<NB, 128, 0, stream>>>(xcur, x0, nmask, (float*)d_out);
}
